// VisionTemporalAttention_41832981463264
// MI455X (gfx1250) — matrix operand order repaired and checked by reading, not compiled, not run on hardware
//
#include <hip/hip_runtime.h>
#include <hip/hip_bf16.h>

typedef unsigned int       u32;
typedef unsigned short     u16;
typedef unsigned long long u64;
typedef __attribute__((ext_vector_type(16))) __bf16 v16bf;
typedef __attribute__((ext_vector_type(8)))  float  v8f;
typedef __attribute__((ext_vector_type(4)))  u32    u32x4;

union Frag16 { u32 u[8]; v16bf v; };

// bf16 pack via hip_bf16 API: lowers to v_cvt_pk_bf16_f32 when the target has it
__device__ __forceinline__ u16 f2bf(float x) {
  union { __hip_bfloat16 h; u16 u; } cv;
  cv.h = __float2bfloat16(x);
  return cv.u;
}
__device__ __forceinline__ u32 pack2bf(float lo, float hi) {
  union { __hip_bfloat162 h; u32 u; } cv;
  cv.h = __float22bfloat162_rn(make_float2(lo, hi));
  return cv.u;
}
__device__ __forceinline__ u64 pack4bf(float4 f) {
  return (u64)pack2bf(f.x, f.y) | ((u64)pack2bf(f.z, f.w) << 32);
}
__device__ __forceinline__ float bflo(u32 w) { return __uint_as_float(w << 16); }
__device__ __forceinline__ float bfhi(u32 w) { return __uint_as_float(w & 0xFFFF0000u); }

// ---------------------------------------------------------------------------
// Tiled GEMM:  C[M,N] = A[M,K] @ B[N,K]^T + bias[N]    (bf16 WMMA, f32 accum)
// A may be fp32 or bf16; C may be written fp32 or bf16.
// Block tile 128x64, K-step 32.  8 waves, each computes a 32x32 sub-tile.
// b128 global loads, register double-buffered against the WMMAs.
// ---------------------------------------------------------------------------
#define GBM 128
#define GBN 64
#define GBK 32
#define GPAD 36   // padded LDS row stride (u16): 72B rows spread banks

template <bool A_BF16, bool OUT_BF16>
__global__ __launch_bounds__(256)
void gemm_bias_wmma(const void* __restrict__ Av,
                    const float* __restrict__ Bw,
                    const float* __restrict__ bias,
                    void* __restrict__ Cv,
                    int M, int N, int K)
{
  __shared__ __align__(16) u16 As[GBM][GPAD];
  __shared__ __align__(16) u16 Bs[GBN][GPAD];

  const float* Af = (const float*)Av;
  const u16*   Ah = (const u16*)Av;
  float* Cf = (float*)Cv;
  u16*   Ch = (u16*)Cv;

  const int tid      = threadIdx.x;
  const int wave     = tid >> 5;
  const int lane     = tid & 31;
  const int lane16   = lane & 15;
  const int laneHalf = lane >> 4;
  const int wr = wave & 3;      // 4 wave rows  -> 128 rows
  const int wc = wave >> 2;     // 2 wave cols  -> 64 cols
  const long rowBase = (long)blockIdx.y * GBM;
  const long colBase = (long)blockIdx.x * GBN;

  // per-thread staging coordinates: 8 chunks of 4 elements per 32-wide row
  const int ar = tid >> 3, ac4 = (tid & 7) * 4;

  v8f acc[2][2];
#pragma unroll
  for (int i = 0; i < 2; ++i)
#pragma unroll
    for (int j = 0; j < 2; ++j)
#pragma unroll
      for (int r = 0; r < 8; ++r) acc[i][j][r] = 0.f;

  float4 aRegF[4];
  u64    aRegH[4];
  float4 bReg[2];
#pragma unroll
  for (int i = 0; i < 4; ++i) {
    if constexpr (A_BF16) aRegH[i] = *(const u64*)&Ah[(rowBase + ar + i * 32) * K + ac4];
    else                  aRegF[i] = *(const float4*)&Af[(rowBase + ar + i * 32) * K + ac4];
  }
#pragma unroll
  for (int i = 0; i < 2; ++i)
    bReg[i] = *(const float4*)&Bw[(colBase + ar + i * 32) * K + ac4];

  for (int k0 = 0; k0 < K; k0 += GBK) {
#pragma unroll
    for (int i = 0; i < 4; ++i) {
      if constexpr (A_BF16) *(u64*)&As[ar + i * 32][ac4] = aRegH[i];
      else                  *(u64*)&As[ar + i * 32][ac4] = pack4bf(aRegF[i]);
    }
#pragma unroll
    for (int i = 0; i < 2; ++i)
      *(u64*)&Bs[ar + i * 32][ac4] = pack4bf(bReg[i]);
    __syncthreads();

    if (k0 + GBK < K) {  // prefetch next tiles into registers (overlaps WMMA)
      const int kn = k0 + GBK;
#pragma unroll
      for (int i = 0; i < 4; ++i) {
        if constexpr (A_BF16) aRegH[i] = *(const u64*)&Ah[(rowBase + ar + i * 32) * K + kn + ac4];
        else                  aRegF[i] = *(const float4*)&Af[(rowBase + ar + i * 32) * K + kn + ac4];
      }
#pragma unroll
      for (int i = 0; i < 2; ++i)
        bReg[i] = *(const float4*)&Bw[(colBase + ar + i * 32) * K + kn + ac4];
    }

    Frag16 a[2], b[2];
#pragma unroll
    for (int mt = 0; mt < 2; ++mt) {
      int m = wr * 32 + mt * 16 + lane16;
#pragma unroll
      for (int v = 0; v < 8; ++v) {
        int kk = laneHalf * 16 + v * 2;
        a[mt].u[v] = *(const u32*)&As[m][kk];
      }
    }
#pragma unroll
    for (int nt = 0; nt < 2; ++nt) {
      int n = wc * 32 + nt * 16 + lane16;
#pragma unroll
      for (int v = 0; v < 8; ++v) {
        int kk = laneHalf * 16 + v * 2;
        b[nt].u[v] = *(const u32*)&Bs[n][kk];
      }
    }
#pragma unroll
    for (int mt = 0; mt < 2; ++mt)
#pragma unroll
      for (int nt = 0; nt < 2; ++nt)
        acc[mt][nt] = __builtin_amdgcn_wmma_f32_16x16x32_bf16(
            false, a[mt].v, false, b[nt].v, (short)0, acc[mt][nt], false, false);
    __syncthreads();
  }

  // epilogue: C-layout (VGPR r -> row r + 8*laneHalf, col = lane16)
#pragma unroll
  for (int mt = 0; mt < 2; ++mt) {
#pragma unroll
    for (int nt = 0; nt < 2; ++nt) {
      long col = colBase + wc * 32 + nt * 16 + lane16;
      float bv = bias[col];
#pragma unroll
      for (int r = 0; r < 8; ++r) {
        long row = rowBase + wr * 32 + mt * 16 + r + laneHalf * 8;
        float val = acc[mt][nt][r] + bv;
        if constexpr (OUT_BF16) Ch[row * N + col] = f2bf(val);
        else                    Cf[row * N + col] = val;
      }
    }
  }
}

// ---------------------------------------------------------------------------
// Temporal attention: one workgroup per (s, h).  T=256, D=80, 8 waves,
// each wave owns 2 blocks of 16 query rows.  Two-pass flash-style softmax:
// pass 1 finds the row max with per-lane accumulation only (one cross-lane
// reduction per block); pass 2 recomputes scores from LDS, exponentiates,
// accumulates per-lane denominators and the P@V WMMAs; normalization is
// deferred to the store.  No score storage -> no register spills.
// QKV arrives as bf16 [65536 x 960]; output bf16 [65536 x 320], row = s*T+t.
// ---------------------------------------------------------------------------
#define AT_T   256
#define AT_S1  256
#define AT_HD  80
#define AT_QS  960
#define KQ_STRIDE 104    // 208B rows: gcd(52,64)=4 -> 16-bank walk; pad 80..103
#define VT_STRIDE 264    // padded row stride for V^T tile (u16)
#define PS_STRIDE 36     // padded row stride for P scratch (u16)
#define SM_KS   0
#define SM_QS   (AT_T * KQ_STRIDE * 2)                       //  53248
#define SM_VT   (SM_QS + AT_T * KQ_STRIDE * 2)               // 106496
#define SM_PS   (SM_VT + AT_HD * VT_STRIDE * 2)              // 148736
#define SM_SIZE (SM_PS + 8 * 16 * PS_STRIDE * 2)             // 157952

__global__ __launch_bounds__(256)
void temporal_attn_wmma(const u16* __restrict__ qkv,
                        const float* __restrict__ rpe,
                        u16* __restrict__ aout)
{
  extern __shared__ char smem[];
  u16* Ks = (u16*)(smem + SM_KS);   // [256][104] K rows (rotary applied)
  u16* Qs = (u16*)(smem + SM_QS);   // [256][104] Q rows (rotary + 1/sqrt(80))
  u16* Vt = (u16*)(smem + SM_VT);   // [80][264]  V transposed (d-major)
  u16* Ps = (u16*)(smem + SM_PS);   // [8][16][36] per-wave P scratch

  const int s = blockIdx.x >> 2;
  const int h = blockIdx.x & 3;
  const int tid = threadIdx.x;
  const int baseQ = h * 80, baseK = 320 + h * 80, baseV = 640 + h * 80;
  const float qscale = 0.11180339887498949f;  // 1/sqrt(80)

  // zero the k-dim pad (cols 80..103) so fragment loads need no bounds checks
  for (int i = tid; i < AT_T * 24; i += 256) {
    int t = i / 24, j = i - t * 24;
    Ks[t * KQ_STRIDE + 80 + j] = 0;
    Qs[t * KQ_STRIDE + 80 + j] = 0;
  }

  // cooperative stage: 8-element (16B, b128) bf16 chunks; 40 % 8 == 0 so a
  // chunk never straddles the rotate-half boundary.
  for (int i = tid; i < AT_T * (AT_HD / 8); i += 256) {
    int t  = i / (AT_HD / 8);
    int d8 = (i - t * (AT_HD / 8)) * 8;             // 0,8,...,72
    const u16* rp = qkv + ((long)t * AT_S1 + s) * AT_QS;
    int pd8 = (d8 < 40) ? d8 + 40 : d8 - 40;
    int dh8 = (d8 < 40) ? d8 : d8 - 40;
    float sgn = (d8 < 40) ? -1.f : 1.f;

    u32x4 kv = *(const u32x4*)&rp[baseK + d8];
    u32x4 kp = *(const u32x4*)&rp[baseK + pd8];
    u32x4 qv = *(const u32x4*)&rp[baseQ + d8];
    u32x4 qp = *(const u32x4*)&rp[baseQ + pd8];
    u32x4 vv = *(const u32x4*)&rp[baseV + d8];
    float4 f0 = *(const float4*)&rpe[t * 40 + dh8];
    float4 f1 = *(const float4*)&rpe[t * 40 + dh8 + 4];

    float fr[8] = {f0.x, f0.y, f0.z, f0.w, f1.x, f1.y, f1.z, f1.w};
    float sn[8], cs[8];
#pragma unroll
    for (int j = 0; j < 8; ++j) __sincosf(fr[j], &sn[j], &cs[j]);

    u32x4 ko, qo;
#pragma unroll
    for (int j = 0; j < 4; ++j) {
      float k0f = bflo(kv[j]) * cs[2 * j]     + sgn * bflo(kp[j]) * sn[2 * j];
      float k1f = bfhi(kv[j]) * cs[2 * j + 1] + sgn * bfhi(kp[j]) * sn[2 * j + 1];
      float q0f = (bflo(qv[j]) * cs[2 * j]     + sgn * bflo(qp[j]) * sn[2 * j])     * qscale;
      float q1f = (bfhi(qv[j]) * cs[2 * j + 1] + sgn * bfhi(qp[j]) * sn[2 * j + 1]) * qscale;
      ko[j] = pack2bf(k0f, k1f);
      qo[j] = pack2bf(q0f, q1f);
      // V is already bf16: split the pair into the transposed tile
      Vt[(d8 + 2 * j)     * VT_STRIDE + t] = (u16)(vv[j] & 0xFFFFu);
      Vt[(d8 + 2 * j + 1) * VT_STRIDE + t] = (u16)(vv[j] >> 16);
    }
    *(u32x4*)&Ks[t * KQ_STRIDE + d8] = ko;
    *(u32x4*)&Qs[t * KQ_STRIDE + d8] = qo;
  }
  __syncthreads();

  const int wave = tid >> 5, lane = tid & 31;
  const int lane16 = lane & 15, laneHalf = lane >> 4;
  u16* Pw = Ps + wave * 16 * PS_STRIDE;

  for (int blk = 2 * wave; blk <= 2 * wave + 1; ++blk) {
    const int qb = blk * 16;
    const int tq = qb + lane16;

    // Q A-fragments from LDS (3 k-steps cover D=80 padded to 96)
    Frag16 qa[3];
#pragma unroll
    for (int ks = 0; ks < 3; ++ks) {
#pragma unroll
      for (int v = 0; v < 8; ++v) {
        int kk = ks * 32 + laneHalf * 16 + v * 2;
        qa[ks].u[v] = *(const u32*)&Qs[tq * KQ_STRIDE + kk];
      }
    }

    // ---- pass 1: per-lane row max across all 16 score tiles ----
    float m8[8];
#pragma unroll
    for (int r = 0; r < 8; ++r) m8[r] = -3.0e38f;

    for (int nt = 0; nt < 16; ++nt) {
      v8f c;
#pragma unroll
      for (int r = 0; r < 8; ++r) c[r] = 0.f;
#pragma unroll
      for (int ks = 0; ks < 3; ++ks) {
        Frag16 b;
#pragma unroll
        for (int v = 0; v < 8; ++v) {
          int kk = ks * 32 + laneHalf * 16 + v * 2;
          int n  = nt * 16 + lane16;
          b.u[v] = *(const u32*)&Ks[n * KQ_STRIDE + kk];
        }
        c = __builtin_amdgcn_wmma_f32_16x16x32_bf16(false, qa[ks].v, false, b.v,
                                                    (short)0, c, false, false);
      }
#pragma unroll
      for (int r = 0; r < 8; ++r) m8[r] = fmaxf(m8[r], c[r]);
    }
    // single cross-lane max reduction (row m = r + 8*laneHalf is in one half)
#pragma unroll
    for (int r = 0; r < 8; ++r) {
      float x = m8[r];
      x = fmaxf(x, __shfl_xor(x, 1, 16));
      x = fmaxf(x, __shfl_xor(x, 2, 16));
      x = fmaxf(x, __shfl_xor(x, 4, 16));
      x = fmaxf(x, __shfl_xor(x, 8, 16));
      m8[r] = x;
    }

    // ---- pass 2: recompute S, P = exp(S - max), O += P @ V, lane sums ----
    float sl[8];
#pragma unroll
    for (int r = 0; r < 8; ++r) sl[r] = 0.f;
    v8f o[5];
#pragma unroll
    for (int i = 0; i < 5; ++i)
#pragma unroll
      for (int r = 0; r < 8; ++r) o[i][r] = 0.f;

    for (int kb = 0; kb < 8; ++kb) {        // 32 kv rows per step
#pragma unroll
      for (int half = 0; half < 2; ++half) {
        int nt = kb * 2 + half;
        v8f c;
#pragma unroll
        for (int r = 0; r < 8; ++r) c[r] = 0.f;
#pragma unroll
        for (int ks = 0; ks < 3; ++ks) {
          Frag16 b;
#pragma unroll
          for (int v = 0; v < 8; ++v) {
            int kk = ks * 32 + laneHalf * 16 + v * 2;
            int n  = nt * 16 + lane16;
            b.u[v] = *(const u32*)&Ks[n * KQ_STRIDE + kk];
          }
          c = __builtin_amdgcn_wmma_f32_16x16x32_bf16(false, qa[ks].v, false, b.v,
                                                      (short)0, c, false, false);
        }
#pragma unroll
        for (int r = 0; r < 8; ++r) {
          float p = __expf(c[r] - m8[r]);
          sl[r] += p;
          Pw[(r + laneHalf * 8) * PS_STRIDE + half * 16 + lane16] = f2bf(p);
        }
      }
      // C-layout -> A-layout via per-wave LDS scratch (DS ops are in-order)
      Frag16 pa;
#pragma unroll
      for (int v = 0; v < 8; ++v) {
        int kk = laneHalf * 16 + v * 2;
        pa.u[v] = *(const u32*)&Pw[lane16 * PS_STRIDE + kk];
      }
#pragma unroll
      for (int ntv = 0; ntv < 5; ++ntv) {
        Frag16 b;
#pragma unroll
        for (int v = 0; v < 8; ++v) {
          int kr = kb * 32 + laneHalf * 16 + v * 2;
          int d  = ntv * 16 + lane16;
          b.u[v] = *(const u32*)&Vt[d * VT_STRIDE + kr];
        }
        o[ntv] = __builtin_amdgcn_wmma_f32_16x16x32_bf16(false, pa.v, false, b.v,
                                                         (short)0, o[ntv], false, false);
      }
    }

    // single cross-lane sum reduction for the softmax denominators
#pragma unroll
    for (int r = 0; r < 8; ++r) {
      float e = sl[r];
      e += __shfl_xor(e, 1, 16);
      e += __shfl_xor(e, 2, 16);
      e += __shfl_xor(e, 4, 16);
      e += __shfl_xor(e, 8, 16);
      sl[r] = e;
    }

    // normalize and store bf16; reference output row order is (s*T + t)
#pragma unroll
    for (int r = 0; r < 8; ++r) {
      float inv = 1.0f / sl[r];
      int t = qb + r + laneHalf * 8;
      long orow = (long)s * AT_T + t;
#pragma unroll
      for (int ntv = 0; ntv < 5; ++ntv)
        aout[orow * 320 + h * 80 + ntv * 16 + lane16] = f2bf(o[ntv][r] * inv);
    }
  }
}

// ---------------------------------------------------------------------------
extern "C" void kernel_launch(void* const* d_in, const int* in_sizes, int n_in,
                              void* d_out, int out_size, void* d_ws, size_t ws_size,
                              hipStream_t stream) {
  const float* hidden = (const float*)d_in[0];
  // d_in[1] = cu_seqlens (S1=256 baked in)
  const float* rpe    = (const float*)d_in[2];
  const float* w_qkv  = (const float*)d_in[3];
  const float* b_qkv  = (const float*)d_in[4];
  const float* w_proj = (const float*)d_in[5];
  const float* b_proj = (const float*)d_in[6];
  float* out = (float*)d_out;

  const int SEQ = 65536;
  u16* qkvbuf  = (u16*)d_ws;                       // bf16 [65536 x 960]
  u16* attnbuf = qkvbuf + (size_t)SEQ * 960;       // bf16 [65536 x 320]

  (void)hipFuncSetAttribute(reinterpret_cast<const void*>(temporal_attn_wmma),
                            hipFuncAttributeMaxDynamicSharedMemorySize, SM_SIZE);

  dim3 blk(256);
  gemm_bias_wmma<false, true><<<dim3(960 / GBN, SEQ / GBM), blk, 0, stream>>>(
      hidden, w_qkv, b_qkv, qkvbuf, SEQ, 960, 1280);
  temporal_attn_wmma<<<dim3(1024), blk, SM_SIZE, stream>>>(qkvbuf, rpe, attnbuf);
  gemm_bias_wmma<true, false><<<dim3(1280 / GBN, SEQ / GBM), blk, 0, stream>>>(
      attnbuf, w_proj, b_proj, out, SEQ, 1280, 320);
}
